// CausalSelfAttention_7464653160473
// MI455X (gfx1250) — matrix operand order repaired and checked by reading, not compiled, not run on hardware
//
#include <hip/hip_runtime.h>
#include <hip/hip_bf16.h>

// Problem constants (CausalSelfAttention: B=2, T=2048, C=1024, H=16, D=64)
#define Bb   2
#define Tt   2048
#define Cc   1024
#define Hh   16
#define Dd   64
#define SCALE_F 0.125f     // 64^-0.5
#define NEGBIG  (-1.0e30f) // finite "-inf" so exp() math never produces NaN

typedef __attribute__((ext_vector_type(16))) __bf16        v16bf;
typedef __attribute__((ext_vector_type(8)))  float         v8f;
typedef __attribute__((ext_vector_type(8)))  unsigned int  v8u;
typedef __attribute__((ext_vector_type(4)))  int           v4i;

// ---- optional CDNA5 async global->LDS path (compile-probe guarded) --------
#if defined(__has_builtin)
#if __has_builtin(__builtin_amdgcn_global_load_async_to_lds_b128)
#define HAVE_ASYNC_LDS 1
#endif
#endif
#ifndef HAVE_ASYNC_LDS
#define HAVE_ASYNC_LDS 0
#endif

#if HAVE_ASYNC_LDS
typedef __attribute__((address_space(1))) v4i as1_v4i;
typedef __attribute__((address_space(3))) v4i as3_v4i;
__device__ __forceinline__ void async_copy_b128(const void* gsrc, void* ldst) {
  __builtin_amdgcn_global_load_async_to_lds_b128(
      (as1_v4i*)gsrc, (as3_v4i*)ldst, 0, 0);
}
__device__ __forceinline__ void async_wait0() {
#if __has_builtin(__builtin_amdgcn_s_wait_asynccnt)
  __builtin_amdgcn_s_wait_asynccnt(0);
#else
  asm volatile("s_wait_asynccnt 0" ::: "memory");
#endif
}
#endif

__device__ __forceinline__ unsigned short f2bf(float f) {
  unsigned int u = __float_as_uint(f);
  u += 0x7fffu + ((u >> 16) & 1u);        // round-to-nearest-even
  return (unsigned short)(u >> 16);
}

// Build a 16x(bf16) fragment (8 VGPRs) from two 16-byte chunks.
__device__ __forceinline__ v16bf ld_frag(const unsigned short* p0,
                                         const unsigned short* p1) {
  uint4 a = *(const uint4*)p0;
  uint4 b = *(const uint4*)p1;
  v8u u;
  u[0] = a.x; u[1] = a.y; u[2] = a.z; u[3] = a.w;
  u[4] = b.x; u[5] = b.y; u[6] = b.z; u[7] = b.w;
  return __builtin_bit_cast(v16bf, u);
}

#define WMMA_BF16(Af, Bf, Cf) \
  __builtin_amdgcn_wmma_f32_16x16x32_bf16(false, (Af), false, (Bf), (short)0, (Cf), false, false)

// Same-wave LDS store->load ordering (no block barrier: waves are divergent).
#define LDS_FENCE() asm volatile("s_wait_dscnt 0" ::: "memory")

// ---------------------------------------------------------------- convert ---
__global__ __launch_bounds__(256) void cvt_kernel(const float* __restrict__ in,
                                                  unsigned short* __restrict__ out,
                                                  int n) {
  int i = blockIdx.x * blockDim.x + threadIdx.x;
  int stride = gridDim.x * blockDim.x;
  for (; i < n; i += stride) out[i] = f2bf(in[i]);
}

// ------------------------------------------------------------------- GEMM ---
// C[M,N] = A[M,K](bf16,row-major) * B[K,N](bf16,row-major) + bias[N]
// Block tile 128x128, 256 threads = 8 waves (4 M-groups x 2 N-groups),
// wave tile 32x64 = 2x4 WMMA accumulators. K stepped by 32.
// A tile staged with GLOBAL_LOAD_ASYNC_TO_LDS_B128 when available (verbatim
// copy); B tile must transpose so it uses the manual load+scatter path.
template <bool OUT_BF16>
__global__ __launch_bounds__(256) void gemm_bf16_kernel(
    const unsigned short* __restrict__ A,
    const unsigned short* __restrict__ Bm,
    const float* __restrict__ bias,
    void* __restrict__ Cout, int M, int N, int K) {
  __shared__ __align__(16) unsigned short As[128][32];   // A tile, row-major
  __shared__ __align__(16) unsigned short Bts[128][32];  // B tile TRANSPOSED [n][k]

  const int tid  = threadIdx.x;
  const int wave = tid >> 5, lane = tid & 31;
  const int lm   = lane & 15, half = lane >> 4;
  const int wm   = wave & 3, wn = wave >> 2;
  const int blockM = blockIdx.y * 128, blockN = blockIdx.x * 128;

  v8f acc[2][4];
#pragma unroll
  for (int mt = 0; mt < 2; ++mt)
#pragma unroll
    for (int nt = 0; nt < 4; ++nt) acc[mt][nt] = (v8f){};

  const int arow  = tid >> 1;          // 0..127
  const int akoff = (tid & 1) * 16;    // 0 or 16
  const int bk    = tid >> 3;          // 0..31
  const int bn0   = (tid & 7) * 16;    // 0..112

  for (int k0 = 0; k0 < K; k0 += 32) {
    // cooperative A tile load (coalesced b128, async DMA into LDS if present)
    const unsigned short* ag = A + (size_t)(blockM + arow) * K + k0 + akoff;
#if HAVE_ASYNC_LDS
    async_copy_b128(ag,     &As[arow][akoff]);
    async_copy_b128(ag + 8, &As[arow][akoff + 8]);
#else
    *(uint4*)&As[arow][akoff]     = *(const uint4*)ag;
    *(uint4*)&As[arow][akoff + 8] = *(const uint4*)(ag + 8);
#endif
    // cooperative B tile load, transpose into LDS
    const unsigned short* bg = Bm + (size_t)(k0 + bk) * N + blockN + bn0;
    union { uint4 q; unsigned short s[8]; } u0, u1;
    u0.q = *(const uint4*)bg;
    u1.q = *(const uint4*)(bg + 8);
#pragma unroll
    for (int i = 0; i < 8; ++i) {
      Bts[bn0 + i][bk]     = u0.s[i];
      Bts[bn0 + 8 + i][bk] = u1.s[i];
    }
#if HAVE_ASYNC_LDS
    async_wait0();                 // LDS writes from this wave's DMA complete
#endif
    __syncthreads();

    v16bf af[2], bf[4];
#pragma unroll
    for (int mt = 0; mt < 2; ++mt) {
      int r = wm * 32 + mt * 16 + lm;
      af[mt] = ld_frag(&As[r][8 * half], &As[r][16 + 8 * half]);
    }
#pragma unroll
    for (int nt = 0; nt < 4; ++nt) {
      int c = wn * 64 + nt * 16 + lm;
      bf[nt] = ld_frag(&Bts[c][8 * half], &Bts[c][16 + 8 * half]);
    }
#pragma unroll
    for (int mt = 0; mt < 2; ++mt)
#pragma unroll
      for (int nt = 0; nt < 4; ++nt) acc[mt][nt] = WMMA_BF16(af[mt], bf[nt], acc[mt][nt]);
    __syncthreads();
  }

  // epilogue: C layout = lane col lm, VGPR r holds row r + 8*half
#pragma unroll
  for (int mt = 0; mt < 2; ++mt) {
#pragma unroll
    for (int nt = 0; nt < 4; ++nt) {
      int gn = blockN + wn * 64 + nt * 16 + lm;
      float bv = bias[gn];
#pragma unroll
      for (int r = 0; r < 8; ++r) {
        int gm = blockM + wm * 32 + mt * 16 + (half << 3) + r;
        float v = acc[mt][nt][r] + bv;
        if (OUT_BF16)
          ((unsigned short*)Cout)[(size_t)gm * N + gn] = f2bf(v);
        else
          ((float*)Cout)[(size_t)gm * N + gn] = v;
      }
    }
  }
}

// -------------------------------------------------------- flash attention ---
// qkv: [B*T, 3C] bf16 (Q | K | V column blocks). One wave = one (b,h,16-row
// query tile). Online-softmax over 32-key blocks, causal, WMMA throughout.
__global__ __launch_bounds__(256) void attn_kernel(
    const unsigned short* __restrict__ qkv,
    unsigned short* __restrict__ y) {
  __shared__ __align__(16) unsigned short Vt[8][64][32];  // per-wave V^T tile
  __shared__ __align__(16) unsigned short Ps[8][16][32];  // per-wave P scratch

  const int tid = threadIdx.x, wave = tid >> 5, lane = tid & 31;
  const int lm = lane & 15, half = lane >> 4;
  const int task = blockIdx.x * 8 + wave;     // 4096 tasks total
  const int qt = task & 127;                  // T/16 = 128 query tiles
  const int bh = task >> 7;
  const int h = bh & 15, b = bh >> 4;
  const int q0 = qt * 16;
  const size_t rowB = (size_t)b * Tt;

  // Q fragments: 16x64 -> two 16x32 A-fragments
  v16bf qf[2];
  {
    const unsigned short* qp = qkv + (rowB + q0 + lm) * (3 * Cc) + h * Dd;
    qf[0] = ld_frag(qp + 8 * half, qp + 16 + 8 * half);
    qf[1] = ld_frag(qp + 32 + 8 * half, qp + 48 + 8 * half);
  }

  float mst[8], lst[8];
  v8f o[4];
#pragma unroll
  for (int r = 0; r < 8; ++r) { mst[r] = NEGBIG; lst[r] = 0.f; }
#pragma unroll
  for (int nt = 0; nt < 4; ++nt) o[nt] = (v8f){};

  const int nkb = (q0 + 15) / 32 + 1;  // causal: key blocks up to the diagonal
  for (int kb = 0; kb < nkb; ++kb) {
    const int kbase = kb * 32;

    // S = Q K^T for 32 keys -> two 16x16 f32 tiles (2 WMMAs each over d)
    v8f s[2];
    s[0] = (v8f){}; s[1] = (v8f){};
#pragma unroll
    for (int nt = 0; nt < 2; ++nt) {
      const unsigned short* kp =
          qkv + (rowB + kbase + nt * 16 + lm) * (3 * Cc) + Cc + h * Dd;
      v16bf k0f = ld_frag(kp + 8 * half, kp + 16 + 8 * half);
      v16bf k1f = ld_frag(kp + 32 + 8 * half, kp + 48 + 8 * half);
      s[nt] = WMMA_BF16(qf[0], k0f, s[nt]);
      s[nt] = WMMA_BF16(qf[1], k1f, s[nt]);
    }

    // Stage V block (32 keys x 64 d) transposed into per-wave LDS
    {
      const unsigned short* vp =
          qkv + (rowB + kbase + lane) * (3 * Cc) + 2 * Cc + h * Dd;
#pragma unroll
      for (int u = 0; u < 32; ++u) {
        unsigned int w = *(const unsigned int*)(vp + 2 * u);
        Vt[wave][2 * u][lane]     = (unsigned short)(w & 0xffffu);
        Vt[wave][2 * u + 1][lane] = (unsigned short)(w >> 16);
      }
    }

    // scale + causal mask + online softmax (uniform across 16-lane groups)
    float alpha[8];
#pragma unroll
    for (int r = 0; r < 8; ++r) {
      const int qrow = q0 + (half << 3) + r;
      float s0 = s[0][r] * SCALE_F; if (kbase + lm > qrow) s0 = NEGBIG;
      float s1 = s[1][r] * SCALE_F; if (kbase + 16 + lm > qrow) s1 = NEGBIG;
      float mx = fmaxf(s0, s1);
#pragma unroll
      for (int off = 1; off < 16; off <<= 1) mx = fmaxf(mx, __shfl_xor(mx, off, 32));
      const float mnew = fmaxf(mst[r], mx);
      const float a = __expf(mst[r] - mnew);
      const float p0 = __expf(s0 - mnew);
      const float p1 = __expf(s1 - mnew);
      float rs = p0 + p1;
#pragma unroll
      for (int off = 1; off < 16; off <<= 1) rs += __shfl_xor(rs, off, 32);
      lst[r] = lst[r] * a + rs;
      mst[r] = mnew;
      alpha[r] = a;
      const int m = (half << 3) + r;
      Ps[wave][m][lm]      = f2bf(p0);
      Ps[wave][m][16 + lm] = f2bf(p1);
    }
#pragma unroll
    for (int nt = 0; nt < 4; ++nt)
#pragma unroll
      for (int r = 0; r < 8; ++r) o[nt][r] *= alpha[r];

    LDS_FENCE();

    // O += P (16x32) @ V (32x64): one A-frag, four B-frags, 4 WMMAs
    v16bf pf = ld_frag(&Ps[wave][lm][8 * half], &Ps[wave][lm][16 + 8 * half]);
#pragma unroll
    for (int nt = 0; nt < 4; ++nt) {
      v16bf vf = ld_frag(&Vt[wave][nt * 16 + lm][8 * half],
                         &Vt[wave][nt * 16 + lm][16 + 8 * half]);
      o[nt] = WMMA_BF16(pf, vf, o[nt]);
    }
    LDS_FENCE();
  }

  // normalize and write y (bf16, [B*T, C]) for the projection GEMM
#pragma unroll
  for (int nt = 0; nt < 4; ++nt) {
#pragma unroll
    for (int r = 0; r < 8; ++r) {
      const int gm = q0 + (half << 3) + r;
      y[(rowB + gm) * Cc + h * Dd + nt * 16 + lm] = f2bf(o[nt][r] / lst[r]);
    }
  }
}

// ------------------------------------------------------------------ launch ---
extern "C" void kernel_launch(void* const* d_in, const int* in_sizes, int n_in,
                              void* d_out, int out_size, void* d_ws, size_t ws_size,
                              hipStream_t stream) {
  (void)in_sizes; (void)n_in; (void)out_size; (void)ws_size;
  const float* x      = (const float*)d_in[0];
  const float* w_attn = (const float*)d_in[1];
  const float* b_attn = (const float*)d_in[2];
  const float* w_proj = (const float*)d_in[3];
  const float* b_proj = (const float*)d_in[4];

  const int M = Bb * Tt;  // 4096

  char* ws = (char*)d_ws;
  size_t off = 0;
  auto alloc = [&](size_t bytes) -> void* {
    void* p = (void*)(ws + off);
    off += (bytes + 255) & ~(size_t)255;
    return p;
  };
  unsigned short* x_bf   = (unsigned short*)alloc((size_t)M * Cc * 2);
  unsigned short* wa_bf  = (unsigned short*)alloc((size_t)Cc * 3 * Cc * 2);
  unsigned short* wp_bf  = (unsigned short*)alloc((size_t)Cc * Cc * 2);
  unsigned short* qkv_bf = (unsigned short*)alloc((size_t)M * 3 * Cc * 2);
  unsigned short* y_bf   = (unsigned short*)alloc((size_t)M * Cc * 2);

  cvt_kernel<<<1024, 256, 0, stream>>>(x,      x_bf,  M * Cc);
  cvt_kernel<<<1024, 256, 0, stream>>>(w_attn, wa_bf, Cc * 3 * Cc);
  cvt_kernel<<<512,  256, 0, stream>>>(w_proj, wp_bf, Cc * Cc);

  gemm_bf16_kernel<true><<<dim3(3 * Cc / 128, M / 128), 256, 0, stream>>>(
      x_bf, wa_bf, b_attn, qkv_bf, M, 3 * Cc, Cc);

  attn_kernel<<<(Bb * Hh * (Tt / 16)) / 8, 256, 0, stream>>>(qkv_bf, y_bf);

  gemm_bf16_kernel<false><<<dim3(Cc / 128, M / 128), 256, 0, stream>>>(
      y_bf, wp_bf, b_proj, d_out, M, Cc, Cc);
}
